// HomoGNNEncoder_67078799228983
// MI455X (gfx1250) — matrix operand order repaired and checked by reading, not compiled, not run on hardware
//
#include <hip/hip_runtime.h>

// ---------------------------------------------------------------------------
// GAT 2-layer forward for MI455X (gfx1250, wave32, WMMA).
// GEMM: fp32-accurate via f16 hi/lo split, 3x v_wmma_f32_16x16x32_f16 per tile
//       with two independent accumulator chains.
// Edge softmax/aggregation: native hardware atomics
//       (global_atomic_max_u32 on monotone-uint floats, global_atomic_add_f32).
// ---------------------------------------------------------------------------

typedef __attribute__((ext_vector_type(16))) _Float16 v16h;
typedef __attribute__((ext_vector_type(8)))  float    v8f;

#define HEADS 4
#define FDIM  128
#define NEG_SLOPE 0.2f

// ---- monotone encoding of float into uint so atomicMax(uint) == float max --
__device__ __forceinline__ unsigned fenc(float f) {
    unsigned b = __float_as_uint(f);
    return (b & 0x80000000u) ? ~b : (b | 0x80000000u);
}
__device__ __forceinline__ float fdec(unsigned k) {
    unsigned b = (k & 0x80000000u) ? (k & 0x7FFFFFFFu) : ~k;
    return __uint_as_float(b);
}
// Non-returning device-scope f32 add: GLOBAL_ATOMIC_ADD_F32, TH[0]=0 (no ret)
__device__ __forceinline__ void fatomic_add(float* p, float v) {
    asm volatile("global_atomic_add_f32 %0, %1, off scope:SCOPE_DEV"
                 :: "v"(p), "v"(v) : "memory");
}
__device__ __forceinline__ float lrelu(float x) {
    return x > 0.0f ? x : NEG_SLOPE * x;
}

// ---- split fp32 -> f16 hi + f16 residual -----------------------------------
__global__ void split_x_k(const float* __restrict__ x,
                          _Float16* __restrict__ hi, _Float16* __restrict__ lo,
                          int total) {
    int i = blockIdx.x * blockDim.x + threadIdx.x;
    if (i >= total) return;
    float v = x[i];
    _Float16 h = (_Float16)v;
    hi[i] = h;
    lo[i] = (_Float16)(v - (float)h);
}

// ---- split + transpose W [K=128][N=128] -> Wt [N][K], f16 hi/lo ------------
__global__ void split_w_k(const float* __restrict__ W,
                          _Float16* __restrict__ hi, _Float16* __restrict__ lo) {
    int i = blockIdx.x * blockDim.x + threadIdx.x;
    if (i >= FDIM * FDIM) return;
    int k = i >> 7, n = i & 127;
    float v = W[k * FDIM + n];
    _Float16 h = (_Float16)v;
    hi[n * FDIM + k] = h;
    lo[n * FDIM + k] = (_Float16)(v - (float)h);
}

// ---- GEMM: out[N x 128] = X[N x 128] * Wt^T, f16 hi/lo 3-product scheme ----
// One wave per 16x16 C tile; block = 8 waves = one 16-row x 128-col strip.
__global__ void gemm_wmma_k(const _Float16* __restrict__ xhi,
                            const _Float16* __restrict__ xlo,
                            const _Float16* __restrict__ wthi,
                            const _Float16* __restrict__ wtlo,
                            float* __restrict__ out, int nrows) {
    const int wave = threadIdx.x >> 5;      // 0..7 -> column tile
    const int lane = threadIdx.x & 31;
    const int rowBase = blockIdx.x * 16;
    const int colBase = wave * 16;

    const int m   = lane & 15;
    const int ahk = (lane >> 4) << 3;       // A: lane-half K offset (0 or 8)
    const int bhk = (lane >> 4) << 4;       // 0 or 16; unused, B uses ahk like A

    int arow = rowBase + m;
    if (arow >= nrows) arow = nrows - 1;    // clamp (keeps EXEC all-ones)
    const int ncol = colBase + m;

    union Frag { v16h h; unsigned u[8]; };

    v8f acc0 = {};                          // hi*hi chain
    v8f acc1 = {};                          // correction-term chain
#pragma unroll
    for (int kk = 0; kk < FDIM; kk += 32) {
        Frag ah, al, bh, bl;
#pragma unroll
        for (int v = 0; v < 8; ++v) {
            // A 16x32 f16 layout: VGPR v holds K pair
            int ka = kk + ((v >> 2) << 4) + ahk + ((v & 3) << 1);
            // B 32x16 f16 layout: same K pairing as A
            int kb = kk + ((v >> 2) << 4) + ahk + ((v & 3) << 1);
            ah.u[v] = *(const unsigned*)(xhi  + arow * FDIM + ka);
            al.u[v] = *(const unsigned*)(xlo  + arow * FDIM + ka);
            bh.u[v] = *(const unsigned*)(wthi + ncol * FDIM + kb);
            bl.u[v] = *(const unsigned*)(wtlo + ncol * FDIM + kb);
        }
        acc0 = __builtin_amdgcn_wmma_f32_16x16x32_f16(false, ah.h, false, bh.h,
                                                      (short)0, acc0, false, false);
        acc1 = __builtin_amdgcn_wmma_f32_16x16x32_f16(false, ah.h, false, bl.h,
                                                      (short)0, acc1, false, false);
        acc1 = __builtin_amdgcn_wmma_f32_16x16x32_f16(false, al.h, false, bh.h,
                                                      (short)0, acc1, false, false);
    }

    // C/D layout: VGPR r -> M = r (lanes 0-15) / r+8 (lanes 16-31), N = lane&15
    const int crow0 = rowBase + ((lane >> 4) << 3);
    const int ccol  = colBase + (lane & 15);
#pragma unroll
    for (int r = 0; r < 8; ++r) {
        int row = crow0 + r;
        if (row < nrows) out[row * FDIM + ccol] = acc0[r] + acc1[r];
    }
}

// ---- per-node attention logits: asrc/adst [N][H] ---------------------------
__global__ void alpha_k(const float* __restrict__ h,
                        const float* __restrict__ a_s, const float* __restrict__ a_d,
                        float* __restrict__ asrc, float* __restrict__ adst, int N) {
    int t = blockIdx.x * blockDim.x + threadIdx.x;
    if (t >= N * HEADS) return;
    int n = t >> 2, hh = t & 3;
    const float* hp = h + n * FDIM + hh * 32;
    const float* sp = a_s + hh * 32;
    const float* dp = a_d + hh * 32;
    float ss = 0.0f, dd = 0.0f;
#pragma unroll
    for (int c = 0; c < 32; ++c) { ss += hp[c] * sp[c]; dd += hp[c] * dp[c]; }
    asrc[t] = ss;
    adst[t] = dd;
}

// ---- zero out accumulator / emax / denom -----------------------------------
__global__ void init_k(unsigned* __restrict__ emax, float* __restrict__ denom,
                       float* __restrict__ outb, int N) {
    int i = blockIdx.x * blockDim.x + threadIdx.x;
    int total = N * FDIM;
    if (i < total) outb[i] = 0.0f;
    if (i < N * HEADS) { emax[i] = 0u; denom[i] = 0.0f; }
}

// ---- edge pass 1: segment max of leaky_relu(asrc[s]+adst[d]) ---------------
__global__ void edge_max_k(const int* __restrict__ src, const int* __restrict__ dst,
                           const float* __restrict__ asrc, const float* __restrict__ adst,
                           unsigned* __restrict__ emax, int E0, int N) {
    int e = blockIdx.x * blockDim.x + threadIdx.x;
    if (e >= E0 + N) return;
    int s = (e < E0) ? src[e] : (e - E0);
    int d = (e < E0) ? dst[e] : (e - E0);
    const float4 av = *(const float4*)(asrc + s * HEADS);
    const float4 dv = *(const float4*)(adst + d * HEADS);
    float vals[4] = {av.x + dv.x, av.y + dv.y, av.z + dv.z, av.w + dv.w};
#pragma unroll
    for (int hh = 0; hh < HEADS; ++hh)
        atomicMax(&emax[d * HEADS + hh], fenc(lrelu(vals[hh])));
}

// ---- edge pass 2: denom = segment sum of exp(e - emax[d]) ------------------
__global__ void edge_sum_k(const int* __restrict__ src, const int* __restrict__ dst,
                           const float* __restrict__ asrc, const float* __restrict__ adst,
                           const unsigned* __restrict__ emax, float* __restrict__ denom,
                           int E0, int N) {
    int e = blockIdx.x * blockDim.x + threadIdx.x;
    if (e >= E0 + N) return;
    int s = (e < E0) ? src[e] : (e - E0);
    int d = (e < E0) ? dst[e] : (e - E0);
    const float4 av = *(const float4*)(asrc + s * HEADS);
    const float4 dv = *(const float4*)(adst + d * HEADS);
    float vals[4] = {av.x + dv.x, av.y + dv.y, av.z + dv.z, av.w + dv.w};
#pragma unroll
    for (int hh = 0; hh < HEADS; ++hh) {
        float w = __expf(lrelu(vals[hh]) - fdec(emax[d * HEADS + hh]));
        fatomic_add(&denom[d * HEADS + hh], w);
    }
}

// ---- edge pass 3: out[d] += alpha * h[s] ; one wave per edge ---------------
__global__ void edge_aggr_k(const int* __restrict__ src, const int* __restrict__ dst,
                            const float* __restrict__ asrc, const float* __restrict__ adst,
                            const unsigned* __restrict__ emax, const float* __restrict__ denom,
                            const float* __restrict__ h, float* __restrict__ outb,
                            int E0, int N) {
    int tid  = blockIdx.x * blockDim.x + threadIdx.x;
    int e    = tid >> 5;
    int lane = tid & 31;
    if (e >= E0 + N) return;
    int s = (e < E0) ? src[e] : (e - E0);
    int d = (e < E0) ? dst[e] : (e - E0);
    int head = lane >> 3;                         // 32 channels/head, 4 ch/lane
    float t = lrelu(asrc[s * HEADS + head] + adst[d * HEADS + head]);
    float w = __expf(t - fdec(emax[d * HEADS + head]));
    float alpha = w / denom[d * HEADS + head];
    const float4 hv = *(const float4*)(h + s * FDIM + lane * 4);
    float* o = outb + d * FDIM + lane * 4;
    fatomic_add(o + 0, alpha * hv.x);
    fatomic_add(o + 1, alpha * hv.y);
    fatomic_add(o + 2, alpha * hv.z);
    fatomic_add(o + 3, alpha * hv.w);
}

// ---- bias + relu -> f32 output ---------------------------------------------
__global__ void bias_relu_k(const float* __restrict__ in, const float* __restrict__ bias,
                            float* __restrict__ out, int total) {
    int i = blockIdx.x * blockDim.x + threadIdx.x;
    if (i >= total) return;
    float v = in[i] + bias[i & 127];
    out[i] = v > 0.0f ? v : 0.0f;
}

// ---- bias + relu -> fused f16 hi/lo split (inter-layer activation) ---------
__global__ void bias_relu_split_k(const float* __restrict__ in,
                                  const float* __restrict__ bias,
                                  _Float16* __restrict__ hi, _Float16* __restrict__ lo,
                                  int total) {
    int i = blockIdx.x * blockDim.x + threadIdx.x;
    if (i >= total) return;
    float v = in[i] + bias[i & 127];
    v = v > 0.0f ? v : 0.0f;
    _Float16 h = (_Float16)v;
    hi[i] = h;
    lo[i] = (_Float16)(v - (float)h);
}

// ---------------------------------------------------------------------------
static inline int cdiv(long long a, long long b) { return (int)((a + b - 1) / b); }

struct Ws {
    float*    h;       // [N*128] transformed features
    float*    outb;    // [N*128] aggregation accumulator
    _Float16* xhi;     // [N*128]
    _Float16* xlo;     // [N*128]
    _Float16* wthi;    // [128*128] transposed
    _Float16* wtlo;    // [128*128]
    float*    asrc;    // [N*4]
    float*    adst;    // [N*4]
    unsigned* emax;    // [N*4]
    float*    denom;   // [N*4]
};

// GEMM (from pre-split xhi/xlo) + attention + aggregation -> w.outb
static void run_core(const int* src, const int* dst,
                     const float* W, const float* a_s, const float* a_d,
                     int N, int E0, const Ws& w, hipStream_t stream) {
    const int B = 256;
    const int total = N * FDIM;
    const int Etot = E0 + N;

    split_w_k<<<cdiv(FDIM * FDIM, B), B, 0, stream>>>(W, w.wthi, w.wtlo);
    gemm_wmma_k<<<cdiv(N, 16), B, 0, stream>>>(w.xhi, w.xlo, w.wthi, w.wtlo, w.h, N);
    alpha_k<<<cdiv(N * HEADS, B), B, 0, stream>>>(w.h, a_s, a_d, w.asrc, w.adst, N);
    init_k<<<cdiv(total, B), B, 0, stream>>>(w.emax, w.denom, w.outb, N);
    edge_max_k<<<cdiv(Etot, B), B, 0, stream>>>(src, dst, w.asrc, w.adst, w.emax, E0, N);
    edge_sum_k<<<cdiv(Etot, B), B, 0, stream>>>(src, dst, w.asrc, w.adst, w.emax, w.denom, E0, N);
    edge_aggr_k<<<cdiv((long long)Etot * 32, B), B, 0, stream>>>(
        src, dst, w.asrc, w.adst, w.emax, w.denom, w.h, w.outb, E0, N);
}

extern "C" void kernel_launch(void* const* d_in, const int* in_sizes, int n_in,
                              void* d_out, int out_size, void* d_ws, size_t ws_size,
                              hipStream_t stream) {
    const float* x   = (const float*)d_in[0];
    const int*   ei  = (const int*)  d_in[1];
    const float* W1  = (const float*)d_in[2];
    const float* as1 = (const float*)d_in[3];
    const float* ad1 = (const float*)d_in[4];
    const float* b1  = (const float*)d_in[5];
    const float* W2  = (const float*)d_in[6];
    const float* as2 = (const float*)d_in[7];
    const float* ad2 = (const float*)d_in[8];
    const float* b2  = (const float*)d_in[9];

    const int N  = in_sizes[0] / FDIM;      // 50000
    const int E0 = in_sizes[1] / 2;         // 800000
    const int* src = ei;
    const int* dst = ei + E0;

    // carve workspace (all regions kept 256B-aligned)
    char* p = (char*)d_ws;
    auto take = [&](size_t bytes) {
        char* r = p;
        p += (bytes + 255) & ~(size_t)255;
        return r;
    };
    Ws w;
    w.h     = (float*)   take((size_t)N * FDIM * sizeof(float));
    w.outb  = (float*)   take((size_t)N * FDIM * sizeof(float));
    w.xhi   = (_Float16*)take((size_t)N * FDIM * sizeof(_Float16));
    w.xlo   = (_Float16*)take((size_t)N * FDIM * sizeof(_Float16));
    w.wthi  = (_Float16*)take((size_t)FDIM * FDIM * sizeof(_Float16));
    w.wtlo  = (_Float16*)take((size_t)FDIM * FDIM * sizeof(_Float16));
    w.asrc  = (float*)   take((size_t)N * HEADS * sizeof(float));
    w.adst  = (float*)   take((size_t)N * HEADS * sizeof(float));
    w.emax  = (unsigned*)take((size_t)N * HEADS * sizeof(unsigned));
    w.denom = (float*)   take((size_t)N * HEADS * sizeof(float));

    const int B = 256;
    const int total = N * FDIM;

    // Layer 1
    split_x_k<<<cdiv(total, B), B, 0, stream>>>(x, w.xhi, w.xlo, total);
    run_core(src, dst, W1, as1, ad1, N, E0, w, stream);
    // inter-layer activation: bias+relu fused with f16 hi/lo split (overwrites xhi/xlo)
    bias_relu_split_k<<<cdiv(total, B), B, 0, stream>>>(w.outb, b1, w.xhi, w.xlo, total);

    // Layer 2
    run_core(src, dst, W2, as2, ad2, N, E0, w, stream);
    bias_relu_k<<<cdiv(total, B), B, 0, stream>>>(w.outb, b2, (float*)d_out, total);
}
